// PoseVQVAE_37477884625424
// MI455X (gfx1250) — matrix operand order repaired and checked by reading, not compiled, not run on hardware
//
#include <hip/hip_runtime.h>
#include <hip/hip_bf16.h>
#include <math.h>

// ---------------------------------------------------------------------------
// PoseVQVAE forward on MI455X (gfx1250): f16 WMMA GEMMs with f32 accumulate.
// VQ kernel stages the codebook in LDS via GLOBAL_LOAD_ASYNC_TO_LDS_B128.
// ---------------------------------------------------------------------------

typedef _Float16 h8v  __attribute__((ext_vector_type(8)));
typedef _Float16 v16h __attribute__((ext_vector_type(16)));
typedef float    v8f  __attribute__((ext_vector_type(8)));

static constexpr int BATCH = 65536;
static constexpr int FD    = 267;
static constexpr int HID   = 256;
static constexpr int LAT   = 64;
static constexpr int NEMB  = 1024;

// ---------------- weight prep kernels ----------------

// dst[k*Np + n] = (k<K && n<N) ? src[n*K + k] : 0   (f32 -> f16, transpose to [Kp][Np])
__global__ void wt_convert_kernel(const float* __restrict__ src, _Float16* __restrict__ dst,
                                  int N, int K, int Kp, int Np) {
    int i = blockIdx.x * 256 + threadIdx.x;
    if (i >= Kp * Np) return;
    int k = i / Np, n = i % Np;
    float v = (k < K && n < N) ? src[(size_t)n * K + k] : 0.0f;
    dst[i] = (_Float16)v;
}

// straight f32 -> f16 convert (embed is already [L][K] = contraction-major)
__global__ void cvt_f16_kernel(const float* __restrict__ src, _Float16* __restrict__ dst, int n) {
    int i = blockIdx.x * 256 + threadIdx.x;
    if (i < n) dst[i] = (_Float16)src[i];
}

// colbias[n] = -0.5 * sum_l embed[l][n]^2
__global__ void colbias_kernel(const float* __restrict__ embed, float* __restrict__ colbias) {
    int n = blockIdx.x * 256 + threadIdx.x;
    if (n >= NEMB) return;
    float s = 0.0f;
    #pragma unroll 8
    for (int l = 0; l < LAT; ++l) {
        float e = embed[(size_t)l * NEMB + n];
        s += e * e;
    }
    colbias[n] = -0.5f * s;
}

__global__ void init_kernel(unsigned int* __restrict__ counts, float* __restrict__ loss_sum) {
    int t = blockIdx.x * 256 + threadIdx.x;
    if (t < NEMB) counts[t] = 0u;
    if (t == 0)   loss_sum[0] = 0.0f;
}

// ---------------- WMMA fragment loaders ----------------

// A fragment (16x32 f16): lane holds row (lane&15); koff = (lane<16)?0:8;
// v16h elems 0..7  -> K = kk + koff + j
// v16h elems 8..15 -> K = kk + 16 + koff + j
// MODE 0: concat( x[f32, w=267], c[f32, w=267] ), zero-pad to Kp
// MODE 1: single f16 buffer, leading dim lda_h (K multiple of 32, no pad)
// MODE 2: concat( q16[f16, w=64], c[f32, w=267] ), zero-pad to Kp (boundary 64 is 8-aligned)
template <int MODE>
__device__ inline v16h load_a_frag(const float* __restrict__ xf, const float* __restrict__ cf,
                                   const _Float16* __restrict__ ah,
                                   int row, int kk, int koff, int lda_h) {
    v16h a;
    #pragma unroll
    for (int h = 0; h < 2; ++h) {
        int g0 = kk + (h ? 16 : 0) + koff;   // always a multiple of 8
        if constexpr (MODE == 1) {
            h8v v = *(const h8v*)(ah + (size_t)row * lda_h + g0);
            #pragma unroll
            for (int j = 0; j < 8; ++j) a[h * 8 + j] = v[j];
        } else if constexpr (MODE == 0) {
            #pragma unroll
            for (int j = 0; j < 8; ++j) {
                int g = g0 + j;
                float f = (g < FD) ? xf[(size_t)row * FD + g]
                        : (g < 2 * FD) ? cf[(size_t)row * FD + (g - FD)]
                        : 0.0f;
                a[h * 8 + j] = (_Float16)f;
            }
        } else {  // MODE 2
            if (g0 + 8 <= LAT) {
                h8v v = *(const h8v*)(ah + (size_t)row * LAT + g0);
                #pragma unroll
                for (int j = 0; j < 8; ++j) a[h * 8 + j] = v[j];
            } else {  // g0 >= 64 here (g0 is 8-aligned)
                #pragma unroll
                for (int j = 0; j < 8; ++j) {
                    int g = g0 + j;
                    float f = (g < LAT + FD) ? cf[(size_t)row * FD + (g - LAT)] : 0.0f;
                    a[h * 8 + j] = (_Float16)f;
                }
            }
        }
    }
    return a;
}

// B fragment (32x16 f16): lane = N column; elems 0..7 -> K = k + j, elems 8..15 -> K = k + 16 + j, from Wt[K][Np]
__device__ inline v16h load_b_frag(const _Float16* __restrict__ wt, int k, int Np, int n0) {
    v16h b;
    #pragma unroll
    for (int j = 0; j < 8; ++j) {
        b[j]     = wt[(size_t)(k + j) * Np + n0];
        b[8 + j] = wt[(size_t)(k + 16 + j) * Np + n0];
    }
    return b;
}

// ---------------- main GEMM: Y = act(A @ Wt + bias) ----------------
// block = 256 threads = 8 waves; tile = 128 rows x 64 cols; wave = 16 rows x 64 cols.
template <int MODE, bool RELU, bool W16, bool W32>
__global__ __launch_bounds__(256)
void gemm_wmma_kernel(const float* __restrict__ xf, const float* __restrict__ cf,
                      const _Float16* __restrict__ ah,
                      const _Float16* __restrict__ wt, const float* __restrict__ bias,
                      _Float16* __restrict__ y16, float* __restrict__ y32,
                      int Kp, int Np, int Nreal, int lda_h, int ld16, int ld32) {
    const int wave = threadIdx.x >> 5;
    const int lane = threadIdx.x & 31;
    const int m0   = blockIdx.x * 128 + wave * 16;
    const int n0   = blockIdx.y * 64;
    const int row  = m0 + (lane & 15);
    const int koff = (lane < 16) ? 0 : 8;

    v8f acc[4] = {};

    for (int kk = 0; kk < Kp; kk += 32) {
        v16h a = load_a_frag<MODE>(xf, cf, ah, row, kk, koff, lda_h);
        const int kb = kk + koff;
        #pragma unroll
        for (int t = 0; t < 4; ++t) {
            v16h b = load_b_frag(wt, kb, Np, n0 + 16 * t + (lane & 15));
            acc[t] = __builtin_amdgcn_wmma_f32_16x16x32_f16(
                false, a, false, b, (short)0, acc[t], false, false);
        }
    }

    // C/D layout: VGPR r -> row (m0 + r) for lanes 0-15, (m0 + 8 + r) for lanes 16-31;
    // column = n0 + 16*t + (lane & 15)
    const int colb  = lane & 15;
    const int rbase = m0 + ((lane >> 4) << 3);
    #pragma unroll
    for (int t = 0; t < 4; ++t) {
        int n = n0 + 16 * t + colb;
        if (n >= Nreal) continue;
        float bv = bias[n];
        #pragma unroll
        for (int r = 0; r < 8; ++r) {
            float v = acc[t][r] + bv;
            if (RELU) v = fmaxf(v, 0.0f);
            int m = rbase + r;
            if (W16) y16[(size_t)m * ld16 + n] = (_Float16)v;
            if (W32) y32[(size_t)m * ld32 + n] = v;
        }
    }
}

// ---------------- VQ: fused scores GEMM + per-row argmax ----------------
// score[b][e] = mu[b]·embed[:,e] - 0.5*||embed[:,e]||^2 ; argmin dist == argmax score.
// Codebook is staged into LDS in two 64KB phases with async global->LDS copies
// (ASYNCcnt path), then WMMA B-fragments are gathered from LDS one f16 element at a time.
__global__ __launch_bounds__(256)
void vq_argmax_kernel(const _Float16* __restrict__ mu16, const _Float16* __restrict__ emb16,
                      const float* __restrict__ colbias,
                      int* __restrict__ idx, unsigned int* __restrict__ counts) {
    __shared__ _Float16 smem[64 * 512];          // 64 KB: one phase of emb16 [64][512]

    const int wave = threadIdx.x >> 5;
    const int lane = threadIdx.x & 31;
    const int m0   = blockIdx.x * 128 + wave * 16;
    const int row  = m0 + (lane & 15);
    const int koff = (lane < 16) ? 0 : 8;

    // A fragments for K = 64 (two k-steps), loaded once and reused for all col tiles
    v16h a0 = load_a_frag<1>(nullptr, nullptr, mu16, row, 0, koff, LAT);
    v16h a1 = load_a_frag<1>(nullptr, nullptr, mu16, row, 32, koff, LAT);

    float best[8];
    int   bidx[8];
    #pragma unroll
    for (int r = 0; r < 8; ++r) { best[r] = -3.4e38f; bidx[r] = 0; }

    // LDS byte offset of smem (flat shared address low 32 bits == LDS offset)
    const unsigned lds_base = (unsigned)(uintptr_t)smem;

    for (int phase = 0; phase < 2; ++phase) {
        // ---- async stage: 64 rows x 512 cols (64 KB), 128 wave-slots of 512B ----
        #pragma unroll
        for (int it = 0; it < 16; ++it) {
            int slot = wave * 16 + it;
            int h    = slot * 256 + lane * 8;            // half-index within phase tile
            int r    = h >> 9;                           // row 0..63
            int cc   = h & 511;                          // col within phase
            unsigned long long g =
                (unsigned long long)(uintptr_t)(emb16 + (size_t)r * NEMB + phase * 512 + cc);
            unsigned l = lds_base + (unsigned)(h * 2);
            asm volatile("global_load_async_to_lds_b128 %0, %1, off"
                         :: "v"(l), "v"(g) : "memory");
        }
        asm volatile("s_wait_asynccnt 0" ::: "memory");
        __syncthreads();

        // ---- 32 column tiles of 16 from LDS ----
        for (int nl = 0; nl < 512; nl += 16) {
            v8f acc = {};
            v16h b0, b1;
            {
                const int cn = nl + (lane & 15);
                #pragma unroll
                for (int j = 0; j < 8; ++j) {
                    b0[j] = smem[(koff + j) * 512 + cn]; b0[8 + j] = smem[(16 + koff + j) * 512 + cn];
                    b1[j] = smem[(32 + koff + j) * 512 + cn]; b1[8 + j] = smem[(48 + koff + j) * 512 + cn];
                }
            }
            acc = __builtin_amdgcn_wmma_f32_16x16x32_f16(false, a0, false, b0, (short)0, acc, false, false);
            acc = __builtin_amdgcn_wmma_f32_16x16x32_f16(false, a1, false, b1, (short)0, acc, false, false);
            int n = phase * 512 + nl + (lane & 15);
            float cb = colbias[n];
            #pragma unroll
            for (int r = 0; r < 8; ++r) {
                float v = acc[r] + cb;
                if (v > best[r]) { best[r] = v; bidx[r] = n; }
            }
        }
        __syncthreads();   // protect smem before next phase overwrites it
    }

    // argmax-reduce over the 16 lanes of each half-wave (xor within bits 0..3),
    // tie-break toward smaller index (matches jnp.argmax "first max")
    #pragma unroll
    for (int off = 8; off >= 1; off >>= 1) {
        #pragma unroll
        for (int r = 0; r < 8; ++r) {
            float ov = __shfl_xor(best[r], off, 32);
            int   oi = __shfl_xor(bidx[r], off, 32);
            if (ov > best[r] || (ov == best[r] && oi < bidx[r])) { best[r] = ov; bidx[r] = oi; }
        }
    }
    if ((lane & 15) == 0) {
        const int rbase = m0 + ((lane >> 4) << 3);
        #pragma unroll
        for (int r = 0; r < 8; ++r) {
            idx[rbase + r] = bidx[r];
            atomicAdd(&counts[bidx[r]], 1u);
        }
    }
}

// ---------------- codebook gather + VQ loss ----------------
__global__ __launch_bounds__(256)
void gather_loss_kernel(const int* __restrict__ idx, const float* __restrict__ embed,
                        const float* __restrict__ mu32, _Float16* __restrict__ q16,
                        float* __restrict__ loss_sum) {
    int b = blockIdx.x * 256 + threadIdx.x;
    int i = idx[b];
    float s = 0.0f;
    #pragma unroll 4
    for (int l = 0; l < LAT; ++l) {
        float q = embed[(size_t)l * NEMB + i];
        q16[(size_t)b * LAT + l] = (_Float16)q;
        float d = q - mu32[(size_t)b * LAT + l];
        s += d * d;
    }
    #pragma unroll
    for (int off = 16; off >= 1; off >>= 1) s += __shfl_xor(s, off, 32);
    if ((threadIdx.x & 31) == 0) atomicAdd(loss_sum, s);
}

// ---------------- loss / perplexity finalize ----------------
__global__ void finalize_kernel(const unsigned int* __restrict__ counts,
                                const float* __restrict__ loss_sum,
                                float* __restrict__ out_scalars) {
    __shared__ float sh[NEMB];
    int t = threadIdx.x;
    float p = (float)counts[t] * (1.0f / (float)BATCH);
    sh[t] = p * logf(p + 1e-10f);
    __syncthreads();
    for (int s = NEMB / 2; s > 0; s >>= 1) {
        if (t < s) sh[t] += sh[t + s];
        __syncthreads();
    }
    if (t == 0) {
        out_scalars[0] = loss_sum[0] * (1.0f / ((float)BATCH * (float)LAT));  // loss
        out_scalars[1] = expf(-sh[0]);                                        // perplexity
    }
}

// ---------------- host orchestration ----------------
extern "C" void kernel_launch(void* const* d_in, const int* in_sizes, int n_in,
                              void* d_out, int out_size, void* d_ws, size_t ws_size,
                              hipStream_t stream) {
    (void)in_sizes; (void)n_in; (void)out_size; (void)ws_size;

    const float* x      = (const float*)d_in[0];
    const float* c      = (const float*)d_in[1];
    const float* fc1_w  = (const float*)d_in[2];
    const float* fc1_b  = (const float*)d_in[3];
    const float* fc2_w  = (const float*)d_in[4];
    const float* fc2_b  = (const float*)d_in[5];
    const float* fc3_w  = (const float*)d_in[6];
    const float* fc3_b  = (const float*)d_in[7];
    const float* mu_w   = (const float*)d_in[8];
    const float* mu_b   = (const float*)d_in[9];
    const float* fc4_w  = (const float*)d_in[10];
    const float* fc4_b  = (const float*)d_in[11];
    const float* fc5_w  = (const float*)d_in[12];
    const float* fc5_b  = (const float*)d_in[13];
    const float* fc6_w  = (const float*)d_in[14];
    const float* fc6_b  = (const float*)d_in[15];
    const float* out_w  = (const float*)d_in[16];
    const float* out_b  = (const float*)d_in[17];
    const float* embed  = (const float*)d_in[18];

    float* recon = (float*)d_out;                               // [B, 267]
    float* out_scalars = recon + (size_t)BATCH * FD;            // loss, perplexity

    // -------- workspace layout --------
    char* ws = (char*)d_ws;
    size_t o = 0;
    auto alloc_h = [&](size_t nh) { _Float16* p = (_Float16*)(ws + o); o += nh * 2; return p; };

    _Float16* fc1_wt = alloc_h(544 * 256);   // [Kp=544][N=256]
    _Float16* fc2_wt = alloc_h(256 * 256);
    _Float16* fc3_wt = alloc_h(256 * 256);
    _Float16* mu_wt  = alloc_h(256 * 64);
    _Float16* fc4_wt = alloc_h(352 * 256);   // [Kp=352][N=256]
    _Float16* fc5_wt = alloc_h(256 * 256);
    _Float16* fc6_wt = alloc_h(256 * 256);
    _Float16* out_wt = alloc_h(256 * 320);   // [K=256][Np=320] (267 padded)
    _Float16* emb16  = alloc_h(64 * 1024);   // [L][K] already contraction-major
    o = (o + 15) & ~(size_t)15;
    float*        colbias  = (float*)(ws + o);        o += NEMB * 4;
    unsigned int* counts   = (unsigned int*)(ws + o); o += NEMB * 4;
    float*        loss_sum = (float*)(ws + o);        o += 16;
    int*          idx      = (int*)(ws + o);          o += (size_t)BATCH * 4;
    _Float16* hA   = alloc_h((size_t)BATCH * 256);
    _Float16* hB   = alloc_h((size_t)BATCH * 256);
    _Float16* mu16 = alloc_h((size_t)BATCH * 64);
    _Float16* q16  = alloc_h((size_t)BATCH * 64);
    float* mu32 = (float*)hB;  // alias: hB is dead between fc3 and fc5; mu32 consumed before fc5

    // -------- weight prep (deterministic each call; tiny vs GEMM cost) --------
    init_kernel<<<dim3(5), 256, 0, stream>>>(counts, loss_sum);
    wt_convert_kernel<<<dim3((544 * 256) / 256), 256, 0, stream>>>(fc1_w, fc1_wt, 256, 534, 544, 256);
    wt_convert_kernel<<<dim3((256 * 256) / 256), 256, 0, stream>>>(fc2_w, fc2_wt, 256, 256, 256, 256);
    wt_convert_kernel<<<dim3((256 * 256) / 256), 256, 0, stream>>>(fc3_w, fc3_wt, 256, 256, 256, 256);
    wt_convert_kernel<<<dim3((256 * 64)  / 256), 256, 0, stream>>>(mu_w,  mu_wt,  64,  256, 256, 64);
    wt_convert_kernel<<<dim3((352 * 256) / 256), 256, 0, stream>>>(fc4_w, fc4_wt, 256, 331, 352, 256);
    wt_convert_kernel<<<dim3((256 * 256) / 256), 256, 0, stream>>>(fc5_w, fc5_wt, 256, 256, 256, 256);
    wt_convert_kernel<<<dim3((256 * 256) / 256), 256, 0, stream>>>(fc6_w, fc6_wt, 256, 256, 256, 256);
    wt_convert_kernel<<<dim3((256 * 320) / 256), 256, 0, stream>>>(out_w, out_wt, 267, 256, 256, 320);
    cvt_f16_kernel<<<dim3((64 * 1024) / 256), 256, 0, stream>>>(embed, emb16, 64 * 1024);
    colbias_kernel<<<dim3(NEMB / 256), 256, 0, stream>>>(embed, colbias);

    const dim3 blk(256);
    const dim3 gH(BATCH / 128, 256 / 64);   // N=256 layers
    const dim3 gL(BATCH / 128, 1);          // N=64 layer
    const dim3 gO(BATCH / 128, 320 / 64);   // output layer (Np=320)

    // -------- encoder --------
    gemm_wmma_kernel<0, true, true, false><<<gH, blk, 0, stream>>>(
        x, c, nullptr, fc1_wt, fc1_b, hA, nullptr, 544, 256, 256, 0, 256, 0);
    gemm_wmma_kernel<1, true, true, false><<<gH, blk, 0, stream>>>(
        nullptr, nullptr, hA, fc2_wt, fc2_b, hB, nullptr, 256, 256, 256, 256, 256, 0);
    gemm_wmma_kernel<1, true, true, false><<<gH, blk, 0, stream>>>(
        nullptr, nullptr, hB, fc3_wt, fc3_b, hA, nullptr, 256, 256, 256, 256, 256, 0);
    gemm_wmma_kernel<1, false, true, true><<<gL, blk, 0, stream>>>(
        nullptr, nullptr, hA, mu_wt, mu_b, mu16, mu32, 256, 64, 64, 256, 64, 64);

    // -------- vector quantizer --------
    vq_argmax_kernel<<<dim3(BATCH / 128), blk, 0, stream>>>(mu16, emb16, colbias, idx, counts);
    gather_loss_kernel<<<dim3(BATCH / 256), blk, 0, stream>>>(idx, embed, mu32, q16, loss_sum);

    // -------- decoder --------
    gemm_wmma_kernel<2, true, true, false><<<gH, blk, 0, stream>>>(
        nullptr, c, q16, fc4_wt, fc4_b, hA, nullptr, 352, 256, 256, 64, 256, 0);
    gemm_wmma_kernel<1, true, true, false><<<gH, blk, 0, stream>>>(
        nullptr, nullptr, hA, fc5_wt, fc5_b, hB, nullptr, 256, 256, 256, 256, 256, 0);
    gemm_wmma_kernel<1, true, true, false><<<gH, blk, 0, stream>>>(
        nullptr, nullptr, hB, fc6_wt, fc6_b, hA, nullptr, 256, 256, 256, 256, 256, 0);
    gemm_wmma_kernel<1, false, false, true><<<gO, blk, 0, stream>>>(
        nullptr, nullptr, hA, out_wt, out_b, nullptr, recon, 256, 320, FD, 256, 0, FD);

    // -------- scalars --------
    finalize_kernel<<<dim3(1), dim3(NEMB), 0, stream>>>(counts, loss_sum, out_scalars);
}
